// CascadeMemoryAttention_32263794327810
// MI455X (gfx1250) — matrix operand order repaired and checked by reading, not compiled, not run on hardware
//
#include <hip/hip_runtime.h>
#include <hip/hip_bf16.h>

// ---------------- problem constants (from reference) ----------------
#define Bn 2
#define Tn 2048
#define Cn 1024
#define Mn 256
#define Sn (Tn + 2 * Mn)   // 2560
#define Hn 16
#define Dn 64
#define GATE_REG 0.01f

typedef __bf16 bf16;
typedef __attribute__((ext_vector_type(16))) __bf16 v16bf;
typedef __attribute__((ext_vector_type(8)))  float  v8f;
typedef __attribute__((ext_vector_type(4)))  int    v4i;

union Frag {
    v16bf v;
    uint4 q[2];
    bf16  e[16];
};

__device__ __forceinline__ v8f wmma_bf16(const Frag& a, const Frag& b, v8f c) {
    // (neg_a, A, neg_b, B, c_mod, C, reuse_a, reuse_b)
    return __builtin_amdgcn_wmma_f32_16x16x32_bf16(false, a.v, false, b.v,
                                                   (short)0, c, false, false);
}

// --------- CDNA5 async global->LDS path (guarded; sync fallback otherwise) ----------
#if defined(__gfx1250__) && __has_builtin(__builtin_amdgcn_global_load_async_to_lds_b128) && \
    __has_builtin(__builtin_amdgcn_s_wait_asynccnt)
#define HAS_ASYNC 1
#else
#define HAS_ASYNC 0
#endif

typedef __attribute__((address_space(1))) v4i gas_v4i;
typedef __attribute__((address_space(3))) v4i las_v4i;

__device__ __forceinline__ void cp16_g2l(const bf16* g, bf16* l) {
#if HAS_ASYNC
    __builtin_amdgcn_global_load_async_to_lds_b128(
        (gas_v4i*)(g), (las_v4i*)(l), 0, 0);
#else
    *(uint4*)l = *(const uint4*)g;
#endif
}

template <int N>
__device__ __forceinline__ void wait_async() {
#if HAS_ASYNC
    __builtin_amdgcn_s_wait_asynccnt(N);
#endif
}

// ---------------- elementwise cast f32 -> bf16 ----------------
__global__ void cast_f32_bf16(const float* __restrict__ src, bf16* __restrict__ dst, int n) {
    int i = blockIdx.x * 256 + threadIdx.x;
    if (i < n) dst[i] = (bf16)src[i];
}

// ---------------- cast + transpose: W[R][Ccols] f32 -> Wt[Ccols][R] bf16 ----------------
__global__ __launch_bounds__(256)
void cast_transpose_k(const float* __restrict__ W, bf16* __restrict__ Wt, int R, int Ccols) {
    __shared__ float tile[32][33];
    int r0 = blockIdx.y * 32, c0 = blockIdx.x * 32;
    int tx = threadIdx.x & 31, ty = threadIdx.x >> 5;   // 32 x 8
#pragma unroll
    for (int i = ty; i < 32; i += 8)
        tile[i][tx] = W[(long)(r0 + i) * Ccols + c0 + tx];
    __syncthreads();
#pragma unroll
    for (int i = ty; i < 32; i += 8)
        Wt[(long)(c0 + i) * R + r0 + tx] = (bf16)tile[tx][i];
}

// ---------------- batched GEMM: C[b] = A[b] (bf16) @ Wt^T (bf16) ----------------
// Wt is pre-transposed [N][K]. Tile 128x128, K-step 32, double-buffered async LDS.
// OUTMODE: 0 = bf16 row-major, 1 = f32 row-major, 2 = bf16 transposed (col*ldc+row).
template <int OUTMODE>
__global__ __launch_bounds__(256)
void gemm_bf16_k(const bf16* __restrict__ A, const bf16* __restrict__ Wt,
                 float* __restrict__ Cf, bf16* __restrict__ Cb,
                 int Mrows, int N, int K, long aBatch, long cBatch, long ldc) {
    __shared__ __align__(16) bf16 Asm[2][128 * 32];   // [row m][k]
    __shared__ __align__(16) bf16 Bsm[2][128 * 32];   // [row n][k]

    const int tid  = threadIdx.x;
    const int lane = tid & 31;
    const int wave = tid >> 5;
    const int b    = blockIdx.z;
    const int m0   = blockIdx.y * 128;
    const int n0   = blockIdx.x * 128;
    const bf16* Ab = A + (long)b * aBatch;

    const int wm = (wave & 3) * 32;
    const int wn = (wave >> 2) * 64;

    v8f acc[2][4];
#pragma unroll
    for (int i = 0; i < 2; ++i)
#pragma unroll
        for (int j = 0; j < 4; ++j)
#pragma unroll
            for (int e = 0; e < 8; ++e) acc[i][j][e] = 0.f;

    // per-thread staging addresses: row = tid>>1 (0..127), cs = (tid&1)*16
    const int srow = tid >> 1;
    const int scs  = (tid & 1) * 16;

    auto issue = [&](int buf, int k0) {
        const bf16* ga = Ab + (long)(m0 + srow) * K + k0 + scs;
        const bf16* gw = Wt + (long)(n0 + srow) * K + k0 + scs;
        bf16* la = &Asm[buf][srow * 32 + scs];
        bf16* lb = &Bsm[buf][srow * 32 + scs];
        cp16_g2l(ga, la);
        cp16_g2l(ga + 8, la + 8);
        cp16_g2l(gw, lb);
        cp16_g2l(gw + 8, lb + 8);
    };

    const int KT = K >> 5;
    issue(0, 0);
    for (int kt = 0; kt < KT; ++kt) {
        const int cur = kt & 1;
        if (kt + 1 < KT) {
            issue(cur ^ 1, (kt + 1) << 5);
            wait_async<4>();   // current tile's 4 async copies completed (in-order)
        } else {
            wait_async<0>();
        }
        __syncthreads();

        Frag af[2], bfr[4];
#pragma unroll
        for (int mi = 0; mi < 2; ++mi) {
            // A operand: lane<16 holds K{0..7,16..23} of row (lane&15); lane>=16 K{8..15,24..31}
            int row = wm + mi * 16 + (lane & 15);
            int k8  = (lane >> 4) * 8;
            af[mi].q[0] = *(const uint4*)&Asm[cur][row * 32 + k8];
            af[mi].q[1] = *(const uint4*)&Asm[cur][row * 32 + 16 + k8];
        }
#pragma unroll
        for (int ni = 0; ni < 4; ++ni) {
            // B operand: lane = column; lanes 0..15 K{0..7,16..23}, lanes 16..31 K{8..15,24..31}
            int col = wn + ni * 16 + (lane & 15);
            int kh  = (lane >> 4) * 8;
            bfr[ni].q[0] = *(const uint4*)&Bsm[cur][col * 32 + kh];
            bfr[ni].q[1] = *(const uint4*)&Bsm[cur][col * 32 + 16 + kh];
        }
#pragma unroll
        for (int mi = 0; mi < 2; ++mi)
#pragma unroll
            for (int ni = 0; ni < 4; ++ni)
                acc[mi][ni] = wmma_bf16(af[mi], bfr[ni], acc[mi][ni]);
        __syncthreads();   // done reading this buffer before it is re-filled
    }

    // epilogue: C layout -> row = vi + (lane>=16 ? 8 : 0), col = lane&15
    const int hi8 = (lane >> 4) * 8;
#pragma unroll
    for (int mi = 0; mi < 2; ++mi)
#pragma unroll
        for (int ni = 0; ni < 4; ++ni)
#pragma unroll
            for (int vi = 0; vi < 8; ++vi) {
                int   row = m0 + wm + mi * 16 + vi + hi8;
                int   col = n0 + wn + ni * 16 + (lane & 15);
                float v   = acc[mi][ni][vi];
                if (OUTMODE == 1)      Cf[(long)b * cBatch + (long)row * ldc + col] = v;
                else if (OUTMODE == 0) Cb[(long)b * cBatch + (long)row * ldc + col] = (bf16)v;
                else                   Cb[(long)b * cBatch + (long)col * ldc + row] = (bf16)v;
            }
}

// ---------------- flash attention with chunk/memory split + gating ----------------
// grid: (Tn/128, Hn, Bn); block 256. Each wave owns 16 Q rows, full d=64.
// V comes pre-transposed: Vt[B][C][S].
__global__ __launch_bounds__(256)
void flash_attn_k(const bf16* __restrict__ Q, const bf16* __restrict__ Kb,
                  const bf16* __restrict__ Vt, const float* __restrict__ G,
                  float* __restrict__ Y) {
    __shared__ __align__(16) bf16 Ksm[32 * 64];        // [key][d]
    __shared__ __align__(16) bf16 Vsm[64 * 32];        // [d][key]
    __shared__ __align__(16) bf16 Psm[8][16 * 32];     // per-wave P tile

    const int tid  = threadIdx.x;
    const int lane = tid & 31;
    const int wave = tid >> 5;
    const int b    = blockIdx.z;
    const int h    = blockIdx.y;
    const int q0   = blockIdx.x * 128;
    const int qb   = q0 + wave * 16;
    const int hi8  = (lane >> 4) * 8;
    const float scale = 0.125f;   // 1/sqrt(64)

    // Q fragments for the whole kernel (K-dim 64 -> two 32-chunks)
    Frag qf[2];
    {
        int t = qb + (lane & 15);
        const bf16* qrow = Q + (long)(b * Tn + t) * Cn + h * Dn;
        int k8 = (lane >> 4) * 8;
#pragma unroll
        for (int kc = 0; kc < 2; ++kc) {
            qf[kc].q[0] = *(const uint4*)(qrow + kc * 32 + k8);
            qf[kc].q[1] = *(const uint4*)(qrow + kc * 32 + 16 + k8);
        }
    }

    float m_run[8], l_run[8];
    v8f accC[4], accM[4];   // chunk vs memory accumulators (shared softmax denom)
#pragma unroll
    for (int i = 0; i < 8; ++i) { m_run[i] = -3.0e38f; l_run[i] = 0.f; }
#pragma unroll
    for (int g2 = 0; g2 < 4; ++g2)
#pragma unroll
        for (int e = 0; e < 8; ++e) { accC[g2][e] = 0.f; accM[g2][e] = 0.f; }

    // per-thread staging addresses (one async B128 each for K and V)
    const int kkey = tid >> 3, kd8 = (tid & 7) * 8;    // K: [32 keys][64 d]
    const int vdd  = tid >> 2, vs8 = (tid & 3) * 8;    // V: [64 d][32 keys]

    const int nbc = blockIdx.x * 4 + 4;     // causal chunk key-blocks of 32
    const int nbm = (2 * Mn) / 32;          // 16 memory key-blocks
    for (int it = 0; it < nbc + nbm; ++it) {
        const bool isMem = (it >= nbc);
        const int  sblk  = isMem ? (Tn + (it - nbc) * 32) : it * 32;

        __syncthreads();
        {   // stage K [32][64] and V [64][32] -- pure contiguous async copies
            const bf16* ksrc = Kb + (long)(b * Sn + sblk + kkey) * Cn + h * Dn + kd8;
            const bf16* vsrc = Vt + ((long)b * Cn + h * Dn + vdd) * Sn + sblk + vs8;
            cp16_g2l(ksrc, &Ksm[kkey * 64 + kd8]);
            cp16_g2l(vsrc, &Vsm[vdd * 32 + vs8]);
            wait_async<0>();
        }
        __syncthreads();

        // scores = Q @ K^T for two 16-key groups
        v8f sc[2];
#pragma unroll
        for (int g2 = 0; g2 < 2; ++g2) {
#pragma unroll
            for (int e = 0; e < 8; ++e) sc[g2][e] = 0.f;
#pragma unroll
            for (int kc = 0; kc < 2; ++kc) {
                Frag kf;
                int key = g2 * 16 + (lane & 15);
                int off = kc * 32 + (lane >> 4) * 8;
                kf.q[0] = *(const uint4*)&Ksm[key * 64 + off];
                kf.q[1] = *(const uint4*)&Ksm[key * 64 + off + 16];
                sc[g2] = wmma_bf16(qf[kc], kf, sc[g2]);
            }
        }

        // scale + causal mask + per-row block max (half-wave shfl reduction)
        float m_blk[8];
#pragma unroll
        for (int vi = 0; vi < 8; ++vi) {
            int   t  = qb + vi + hi8;
            float s0 = sc[0][vi] * scale;
            float s1 = sc[1][vi] * scale;
            if (!isMem) {
                int c0 = sblk + (lane & 15);
                if (c0 > t)      s0 = -3.0e38f;
                if (c0 + 16 > t) s1 = -3.0e38f;
            }
            sc[0][vi] = s0;
            sc[1][vi] = s1;
            float m = fmaxf(s0, s1);
#pragma unroll
            for (int x = 8; x >= 1; x >>= 1) m = fmaxf(m, __shfl_xor(m, x, 32));
            m_blk[vi] = m;
        }

        // online softmax update
        float corr[8];
#pragma unroll
        for (int vi = 0; vi < 8; ++vi) {
            float mn  = fmaxf(m_run[vi], m_blk[vi]);
            corr[vi]  = __expf(m_run[vi] - mn);
            m_run[vi] = mn;
            float p0 = (sc[0][vi] > -1.0e37f) ? __expf(sc[0][vi] - mn) : 0.f;
            float p1 = (sc[1][vi] > -1.0e37f) ? __expf(sc[1][vi] - mn) : 0.f;
            sc[0][vi] = p0;
            sc[1][vi] = p1;
            float l = p0 + p1;
#pragma unroll
            for (int x = 8; x >= 1; x >>= 1) l += __shfl_xor(l, x, 32);
            l_run[vi] = l_run[vi] * corr[vi] + l;
        }
#pragma unroll
        for (int g2 = 0; g2 < 4; ++g2)
#pragma unroll
            for (int vi = 0; vi < 8; ++vi) {
                accC[g2][vi] *= corr[vi];
                accM[g2][vi] *= corr[vi];
            }

        // P -> per-wave LDS bounce, reload in A-operand layout
#pragma unroll
        for (int vi = 0; vi < 8; ++vi) {
            int row = vi + hi8;
            Psm[wave][row * 32 + (lane & 15)]      = (bf16)sc[0][vi];
            Psm[wave][row * 32 + 16 + (lane & 15)] = (bf16)sc[1][vi];
        }
        Frag pf;
        {
            int row = lane & 15;
            int k8  = (lane >> 4) * 8;
            pf.q[0] = *(const uint4*)&Psm[wave][row * 32 + k8];
            pf.q[1] = *(const uint4*)&Psm[wave][row * 32 + 16 + k8];
        }

        // P @ V into chunk or memory accumulator (uniform branch)
#pragma unroll
        for (int g2 = 0; g2 < 4; ++g2) {
            Frag vf;
            int dcol = g2 * 16 + (lane & 15);
            int koff = (lane >> 4) * 8;
            vf.q[0] = *(const uint4*)&Vsm[dcol * 32 + koff];
            vf.q[1] = *(const uint4*)&Vsm[dcol * 32 + 16 + koff];
            if (isMem) accM[g2] = wmma_bf16(pf, vf, accM[g2]);
            else       accC[g2] = wmma_bf16(pf, vf, accC[g2]);
        }
    }

    // epilogue: y = (y_chunk + g * y_mem) / l
#pragma unroll
    for (int vi = 0; vi < 8; ++vi) {
        int   t   = qb + vi + hi8;
        float gv  = G[(long)(b * Tn + t) * Hn + h];
        float inv = 1.0f / l_run[vi];
#pragma unroll
        for (int g2 = 0; g2 < 4; ++g2) {
            float o = (accC[g2][vi] + gv * accM[g2][vi]) * inv;
            Y[(long)(b * Tn + t) * Cn + h * Dn + g2 * 16 + (lane & 15)] = o;
        }
    }
}

// ---------------- gate: g = sigmoid(q @ gate_w + gate_b), plus L1 loss ----------------
__global__ __launch_bounds__(256)
void gate_k(const bf16* __restrict__ Q, const float* __restrict__ GW,
            const float* __restrict__ GB, float* __restrict__ G,
            float* __restrict__ loss) {
    __shared__ float red[256];
    int idx = blockIdx.x * 256 + threadIdx.x;    // over B*T*H
    int hh  = idx % Hn;
    int bt  = idx / Hn;
    const bf16* qrow = Q + (long)bt * Cn;
    float s = 0.f;
    for (int c = 0; c < Cn; ++c) s += (float)qrow[c] * GW[c * Hn + hh];
    s += GB[hh];
    float gv = 1.f / (1.f + __expf(-s));
    G[idx]   = gv;
    red[threadIdx.x] = fabsf(gv);
    __syncthreads();
    for (int st = 128; st > 0; st >>= 1) {
        if (threadIdx.x < st) red[threadIdx.x] += red[threadIdx.x + st];
        __syncthreads();
    }
    if (threadIdx.x == 0)
        atomicAdd(loss, red[0] * (GATE_REG / (float)(Bn * Tn * Hn)));
}

// ---------------- causal depthwise conv (K=4) + residual, emit bf16 ----------------
__global__ void conv_k(const float* __restrict__ A, const float* __restrict__ CW,
                       const float* __restrict__ CB, bf16* __restrict__ Out) {
    long idx = (long)blockIdx.x * 256 + threadIdx.x;   // over B*T*C
    int  c   = (int)(idx % Cn);
    int  t   = (int)((idx / Cn) % Tn);
    long base = idx - c - (long)t * Cn;                // b*T*C
    float s = CB[c];
#pragma unroll
    for (int i = 0; i < 4; ++i) {
        int ts = t - 3 + i;
        if (ts >= 0) s += CW[i * Cn + c] * A[base + (long)ts * Cn + c];
    }
    Out[idx] = (bf16)(A[idx] + s);
}

__global__ void zero_loss_k(float* p) { *p = 0.f; }

// ---------------- host side ----------------
extern "C" void kernel_launch(void* const* d_in, const int* in_sizes, int n_in,
                              void* d_out, int out_size, void* d_ws, size_t ws_size,
                              hipStream_t stream) {
    (void)in_sizes; (void)n_in; (void)out_size; (void)ws_size;
    const float* x   = (const float*)d_in[0];
    const float* fm  = (const float*)d_in[1];
    const float* rm  = (const float*)d_in[2];
    const float* w_q = (const float*)d_in[3];
    const float* w_k = (const float*)d_in[4];
    const float* w_v = (const float*)d_in[5];
    const float* w_o = (const float*)d_in[6];
    const float* gw  = (const float*)d_in[7];
    const float* gb  = (const float*)d_in[8];
    const float* cw  = (const float*)d_in[9];
    const float* cb  = (const float*)d_in[10];
    float* out = (float*)d_out;

    // workspace carve-up (all sizes multiples of 256B)
    char* ws = (char*)d_ws;
    auto alloc = [&](size_t bytes) {
        char* p = ws;
        ws += (bytes + 255) & ~(size_t)255;
        return p;
    };
    bf16*  Xc   = (bf16*)alloc((size_t)Bn * Sn * Cn * 2);   // concat(x, fm, rm) bf16
    bf16*  Wqt  = (bf16*)alloc((size_t)Cn * Cn * 2);        // transposed weights [N][K]
    bf16*  Wkt  = (bf16*)alloc((size_t)Cn * Cn * 2);
    bf16*  Wvt  = (bf16*)alloc((size_t)Cn * Cn * 2);
    bf16*  Wot  = (bf16*)alloc((size_t)Cn * Cn * 2);
    bf16*  Qb   = (bf16*)alloc((size_t)Bn * Tn * Cn * 2);
    bf16*  Kb   = (bf16*)alloc((size_t)Bn * Sn * Cn * 2);
    bf16*  Vt   = (bf16*)alloc((size_t)Bn * Cn * Sn * 2);   // V transposed [B][C][S]
    float* Gbuf = (float*)alloc((size_t)Bn * Tn * Hn * 4);
    float* Attn = (float*)alloc((size_t)Bn * Tn * Cn * 4);
    bf16*  A2   = (bf16*)alloc((size_t)Bn * Tn * Cn * 2);

    auto cast = [&](const float* s, bf16* d, int n) {
        cast_f32_bf16<<<dim3((n + 255) / 256), dim3(256), 0, stream>>>(s, d, n);
    };
    for (int b = 0; b < Bn; ++b) {
        cast(x  + (long)b * Tn * Cn, Xc + (long)b * Sn * Cn,                        Tn * Cn);
        cast(fm + (long)b * Mn * Cn, Xc + (long)b * Sn * Cn + (long)Tn * Cn,        Mn * Cn);
        cast(rm + (long)b * Mn * Cn, Xc + (long)b * Sn * Cn + (long)(Tn + Mn) * Cn, Mn * Cn);
    }
    dim3 tgrid(Cn / 32, Cn / 32);
    cast_transpose_k<<<tgrid, dim3(256), 0, stream>>>(w_q, Wqt, Cn, Cn);
    cast_transpose_k<<<tgrid, dim3(256), 0, stream>>>(w_k, Wkt, Cn, Cn);
    cast_transpose_k<<<tgrid, dim3(256), 0, stream>>>(w_v, Wvt, Cn, Cn);
    cast_transpose_k<<<tgrid, dim3(256), 0, stream>>>(w_o, Wot, Cn, Cn);

    // projections (bf16 out): Q uses only the first T rows of each batch of Xc
    gemm_bf16_k<0><<<dim3(Cn / 128, Tn / 128, Bn), dim3(256), 0, stream>>>(
        Xc, Wqt, nullptr, Qb, Tn, Cn, Cn, (long)Sn * Cn, (long)Tn * Cn, Cn);
    gemm_bf16_k<0><<<dim3(Cn / 128, Sn / 128, Bn), dim3(256), 0, stream>>>(
        Xc, Wkt, nullptr, Kb, Sn, Cn, Cn, (long)Sn * Cn, (long)Sn * Cn, Cn);
    gemm_bf16_k<2><<<dim3(Cn / 128, Sn / 128, Bn), dim3(256), 0, stream>>>(
        Xc, Wvt, nullptr, Vt, Sn, Cn, Cn, (long)Sn * Cn, (long)Cn * Sn, Sn);

    // gate + L1 loss (loss scalar lives at d_out[B*T*C])
    float* loss = out + (size_t)Bn * Tn * Cn;
    zero_loss_k<<<dim3(1), dim3(1), 0, stream>>>(loss);
    gate_k<<<dim3((Bn * Tn * Hn) / 256), dim3(256), 0, stream>>>(Qb, gw, gb, Gbuf, loss);

    // fused flash attention with chunk/memory gating
    flash_attn_k<<<dim3(Tn / 128, Hn, Bn), dim3(256), 0, stream>>>(Qb, Kb, Vt, Gbuf, Attn);

    // causal depthwise conv + residual (bf16 out for final GEMM)
    conv_k<<<dim3((Bn * Tn * Cn) / 256), dim3(256), 0, stream>>>(Attn, cw, cb, A2);

    // output projection straight into d_out (f32)
    gemm_bf16_k<1><<<dim3(Cn / 128, Tn / 128, Bn), dim3(256), 0, stream>>>(
        A2, Wot, out, nullptr, Tn, Cn, Cn, (long)Tn * Cn, (long)Tn * Cn, Cn);
}
